// WindowAttentionBlock_37477884625377
// MI455X (gfx1250) — matrix operand order repaired and checked by reading, not compiled, not run on hardware
//
#include <hip/hip_runtime.h>

typedef __attribute__((ext_vector_type(16))) _Float16 v16h;
typedef __attribute__((ext_vector_type(8)))  _Float16 v8h;
typedef __attribute__((ext_vector_type(8)))  float    v8f;

#define D_MODEL 384
#define NH      12
#define HDIM    32
#define NTOK    49
#define NPAD    64
// faithful to reference: scale = 32^-5
#define SCALE_F 2.9802322387695312e-8f
#define LXSTR   392   // 384 + 8 halves pad; 392*2 = 784 bytes, 16-byte aligned rows

union Frag16 { v16h v; v8h h[2]; };

static __device__ __forceinline__ v8f zero_v8f() {
  v8f z;
#pragma unroll
  for (int i = 0; i < 8; ++i) z[i] = 0.f;
  return z;
}

// ---------------------------------------------------------------------------
// Kernel 0: convert all four weight matrices to f16, TRANSPOSED (Wt[c][k]).
// Every WMMA B-fragment becomes one contiguous 32-byte load; the whole f16
// weight set (1.2 MB) stays resident in the 192 MB L2.
// ---------------------------------------------------------------------------
__global__ __launch_bounds__(256) void cvt_weights_kernel(
    const float* __restrict__ Wq, const float* __restrict__ Wk,
    const float* __restrict__ Wv, const float* __restrict__ Wp,
    _Float16* __restrict__ wbuf)
{
  const int sel = blockIdx.y;
  const float* __restrict__ W =
      (sel == 0) ? Wq : (sel == 1) ? Wk : (sel == 2) ? Wv : Wp;
  _Float16* __restrict__ dst = wbuf + (size_t)sel * (D_MODEL * D_MODEL);
  const int idx = blockIdx.x * 256 + threadIdx.x;
  if (idx < D_MODEL * D_MODEL) {
    const int c = idx / D_MODEL, k = idx % D_MODEL;
    dst[idx] = (_Float16)W[(size_t)k * D_MODEL + c];   // transpose
  }
}

// ---------------------------------------------------------------------------
// Kernel 1: Q/K/V projection.  grid = (B, 3), 256 threads (8 waves).
// Rolled K loop + loop-carried B double-buffer + launch_bounds occupancy cap:
// >=4 waves/SIMD so memory waits are hidden across waves, not registers.
// ---------------------------------------------------------------------------
__global__ __launch_bounds__(256, 4) void qkv_proj_kernel(
    const float* __restrict__ x,
    const _Float16* __restrict__ wbuf,
    const float* __restrict__ bq, const float* __restrict__ bk, const float* __restrict__ bv,
    _Float16* __restrict__ qbuf, _Float16* __restrict__ kbuf, _Float16* __restrict__ vtbuf)
{
  __shared__ _Float16 lx[NPAD * LXSTR];
  const int b   = blockIdx.x;
  const int sel = blockIdx.y;
  const _Float16* __restrict__ w16  = wbuf + (size_t)sel * (D_MODEL * D_MODEL);
  const float*    __restrict__ bias = (sel == 0) ? bq : ((sel == 1) ? bk : bv);
  const int tid = threadIdx.x;

  // warm L2 for the f16 weights (gfx1250 global_prefetch path)
  __builtin_prefetch(w16 + ((tid * 577) % (D_MODEL * D_MODEL)), 0, 0);

  // stage x window into LDS as f16 (float4 coalesced loads), zero-pad rows 49..63
  const float4* __restrict__ xb4 =
      (const float4*)(x + (size_t)b * NTOK * D_MODEL);
  for (int i = tid; i < (NTOK * D_MODEL) / 4; i += 256) {
    const float4 f = xb4[i];
    const int e = i * 4;
    _Float16* p = &lx[(e / D_MODEL) * LXSTR + (e % D_MODEL)];
    p[0] = (_Float16)f.x; p[1] = (_Float16)f.y;
    p[2] = (_Float16)f.z; p[3] = (_Float16)f.w;
  }
  for (int i = tid; i < (NPAD - NTOK) * D_MODEL; i += 256) {
    const int r = NTOK + i / D_MODEL, c = i % D_MODEL;
    lx[r * LXSTR + c] = (_Float16)0.f;
  }
  __syncthreads();

  const int lane = tid & 31;
  const int wv   = tid >> 5;     // wave 0..7
  const int r    = lane & 15;
  const int h2   = lane >> 4;    // half-wave 0/1

  for (int it = 0; it < 3; ++it) {
    const int nt = wv * 3 + it;          // column tile 0..23
    const int c  = nt * 16 + r;          // this lane's output column
    v8f acc[4];
#pragma unroll
    for (int mt = 0; mt < 4; ++mt) acc[mt] = zero_v8f();

    // B fragments: contiguous 32B from transposed f16 weights, double-buffered
    // one full iteration (4 WMMAs) ahead.  The kk+64 prefetch may over-read
    // <=128B past this matrix's end; it lands in the next ws buffer (harmless,
    // never consumed).
    const _Float16* __restrict__ wcol = w16 + (size_t)c * D_MODEL + 16 * h2;
    v16h bf_cur = *(const v16h*)(wcol);
    v16h bf_nxt = *(const v16h*)(wcol + 32);
#pragma unroll 1
    for (int kk = 0; kk < D_MODEL; kk += 32) {
      const v16h bf = bf_cur;
      bf_cur = bf_nxt;
      bf_nxt = *(const v16h*)(wcol + kk + 64);
#pragma unroll
      for (int mt = 0; mt < 4; ++mt) {
        const int row = mt * 16 + r;
        Frag16 af;
        af.h[0] = *(const v8h*)&lx[row * LXSTR + kk + 16 * h2];
        af.h[1] = *(const v8h*)&lx[row * LXSTR + kk + 16 * h2 + 8];
        acc[mt] = __builtin_amdgcn_wmma_f32_16x16x32_f16(
            false, af.v, false, bf, (short)0, acc[mt], false, false);
      }
    }

    // branch-free epilogue: resolve layout once
    const int   hh = c >> 5;     // head
    const int   dd = c & 31;     // dim within head
    const float bc = bias[c];
    const size_t base = (size_t)(b * NH + hh) * (NPAD * HDIM);
    _Float16* __restrict__ dst;
    int rstride;
    if (sel == 2)      { dst = vtbuf + base + (size_t)dd * NPAD; rstride = 1; }
    else if (sel == 0) { dst = qbuf  + base + dd;                rstride = HDIM; }
    else               { dst = kbuf  + base + dd;                rstride = HDIM; }
#pragma unroll
    for (int mt = 0; mt < 4; ++mt)
#pragma unroll
      for (int i2 = 0; i2 < 8; ++i2) {
        const int row = mt * 16 + i2 + 8 * h2;
        dst[row * rstride] = (_Float16)(acc[mt][i2] + bc);
      }
  }
}

// ---------------------------------------------------------------------------
// Kernel 2: attention for one (b, head).  grid = B*H, 128 threads (4 waves).
// S = Q K^T * SCALE + mask -> softmax(rows over 49 cols) -> O = W V
// ---------------------------------------------------------------------------
__global__ __launch_bounds__(128) void attn_kernel(
    const _Float16* __restrict__ qbuf, const _Float16* __restrict__ kbuf,
    const _Float16* __restrict__ vtbuf, const float* __restrict__ mask,
    _Float16* __restrict__ obuf)
{
  __shared__ float    sc[NPAD * 68];   // scores, stride 68 floats
  __shared__ _Float16 wp[NPAD * 72];   // softmax weights f16, stride 72 halves (144B)

  const int bh  = blockIdx.x;
  const int b   = bh / NH;
  const int hd  = bh % NH;
  const int tid = threadIdx.x;
  const int lane = tid & 31;
  const int mt   = tid >> 5;          // wave = row tile 0..3
  const int r    = lane & 15;
  const int h2   = lane >> 4;

  const _Float16* __restrict__ qb = qbuf  + (size_t)bh * (NPAD * HDIM);
  const _Float16* __restrict__ kb = kbuf  + (size_t)bh * (NPAD * HDIM);
  const _Float16* __restrict__ vt = vtbuf + (size_t)bh * (NPAD * HDIM);

  // ---- scores: S[mt*16.., nt*16..] (K = 32 = head dim, one WMMA each) ----
  Frag16 af;
  af.h[0] = *(const v8h*)(qb + (mt * 16 + r) * HDIM + 16 * h2);
  af.h[1] = *(const v8h*)(qb + (mt * 16 + r) * HDIM + 16 * h2 + 8);
  v8f acc[4];
#pragma unroll
  for (int nt = 0; nt < 4; ++nt) acc[nt] = zero_v8f();
#pragma unroll
  for (int nt = 0; nt < 4; ++nt) {
    const v16h bf = *(const v16h*)(kb + (nt * 16 + r) * HDIM + 16 * h2);
    acc[nt] = __builtin_amdgcn_wmma_f32_16x16x32_f16(
        false, af.v, false, bf, (short)0, acc[nt], false, false);
  }
  const float* __restrict__ mk = mask + (size_t)b * NTOK * NTOK;
#pragma unroll
  for (int nt = 0; nt < 4; ++nt)
#pragma unroll
    for (int i2 = 0; i2 < 8; ++i2) {
      const int m = mt * 16 + i2 + 8 * h2;
      const int j = nt * 16 + r;
      float v = acc[nt][i2] * SCALE_F;
      if (j < NTOK) { if (m < NTOK) v += mk[m * NTOK + j]; }
      else v = -1e30f;
      sc[m * 68 + j] = v;
    }
  __syncthreads();

  // ---- softmax: one thread per row ----
  if (tid < NPAD) {
    const int row = tid;
    float mx = -1e30f;
    for (int j = 0; j < NTOK; ++j) mx = fmaxf(mx, sc[row * 68 + j]);
    float s = 0.f;
    for (int j = 0; j < NTOK; ++j) {
      const float e = expf(sc[row * 68 + j] - mx);
      sc[row * 68 + j] = e;
      s += e;
    }
    const float inv = 1.f / s;
    for (int j = 0; j < NTOK; ++j) wp[row * 72 + j] = (_Float16)(sc[row * 68 + j] * inv);
    for (int j = NTOK; j < NPAD; ++j) wp[row * 72 + j] = (_Float16)0.f;
  }
  __syncthreads();

  // ---- O = W @ V : rows mt*16.., K=64, N=32 ----
  v8f oacc[2];
#pragma unroll
  for (int nt = 0; nt < 2; ++nt) oacc[nt] = zero_v8f();
#pragma unroll
  for (int kk = 0; kk < NPAD; kk += 32) {
    Frag16 aw;
    aw.h[0] = *(const v8h*)&wp[(mt * 16 + r) * 72 + kk + 16 * h2];
    aw.h[1] = *(const v8h*)&wp[(mt * 16 + r) * 72 + kk + 16 * h2 + 8];
#pragma unroll
    for (int nt = 0; nt < 2; ++nt) {
      // B[k=m][n=d] = V[m][d] = Vt[d][m] -> contiguous in m
      const v16h bf = *(const v16h*)(vt + (size_t)(nt * 16 + r) * NPAD + kk + 16 * h2);
      oacc[nt] = __builtin_amdgcn_wmma_f32_16x16x32_f16(
          false, aw.v, false, bf, (short)0, oacc[nt], false, false);
    }
  }
#pragma unroll
  for (int nt = 0; nt < 2; ++nt)
#pragma unroll
    for (int i2 = 0; i2 < 8; ++i2) {
      const int m = mt * 16 + i2 + 8 * h2;
      if (m < NTOK) {
        const int c = hd * HDIM + nt * 16 + r;
        obuf[((size_t)b * NTOK + m) * D_MODEL + c] = (_Float16)oacc[nt][i2];
      }
    }
}

// ---------------------------------------------------------------------------
// Kernel 3: output projection.  grid = B, 256 threads.
// out[49x384] = O[49x384] @ Wp + bp   (fp32 output)
// ---------------------------------------------------------------------------
__global__ __launch_bounds__(256, 4) void out_proj_kernel(
    const _Float16* __restrict__ obuf, const _Float16* __restrict__ wbuf,
    const float* __restrict__ bp, float* __restrict__ out)
{
  __shared__ _Float16 lo[NPAD * LXSTR];
  const int b   = blockIdx.x;
  const int tid = threadIdx.x;
  const _Float16* __restrict__ w16 = wbuf + (size_t)3 * (D_MODEL * D_MODEL);

  __builtin_prefetch(w16 + ((tid * 577) % (D_MODEL * D_MODEL)), 0, 0);

  // stage O tile (f16) into LDS with 16-byte copies, zero-pad rows 49..63
  const uint4* __restrict__ ob4 =
      (const uint4*)(obuf + (size_t)b * NTOK * D_MODEL);
  for (int i = tid; i < (NTOK * D_MODEL) / 8; i += 256) {
    const int e = i * 8;
    *(uint4*)&lo[(e / D_MODEL) * LXSTR + (e % D_MODEL)] = ob4[i];
  }
  for (int i = tid; i < (NPAD - NTOK) * D_MODEL; i += 256) {
    const int r = NTOK + i / D_MODEL, c = i % D_MODEL;
    lo[r * LXSTR + c] = (_Float16)0.f;
  }
  __syncthreads();

  const int lane = tid & 31;
  const int wv   = tid >> 5;
  const int r    = lane & 15;
  const int h2   = lane >> 4;

  for (int it = 0; it < 3; ++it) {
    const int nt = wv * 3 + it;
    const int c  = nt * 16 + r;
    v8f acc[4];
#pragma unroll
    for (int mt = 0; mt < 4; ++mt) acc[mt] = zero_v8f();

    const _Float16* __restrict__ wcol = w16 + (size_t)c * D_MODEL + 16 * h2;
    v16h bf_cur = *(const v16h*)(wcol);
    v16h bf_nxt = *(const v16h*)(wcol + 32);
#pragma unroll 1
    for (int kk = 0; kk < D_MODEL; kk += 32) {
      const v16h bf = bf_cur;
      bf_cur = bf_nxt;
      bf_nxt = *(const v16h*)(wcol + kk + 64);   // tail over-read stays in d_ws
#pragma unroll
      for (int mt = 0; mt < 4; ++mt) {
        const int row = mt * 16 + r;
        Frag16 af;
        af.h[0] = *(const v8h*)&lo[row * LXSTR + kk + 16 * h2];
        af.h[1] = *(const v8h*)&lo[row * LXSTR + kk + 16 * h2 + 8];
        acc[mt] = __builtin_amdgcn_wmma_f32_16x16x32_f16(
            false, af.v, false, bf, (short)0, acc[mt], false, false);
      }
    }
    const float bc = bp[c];
#pragma unroll
    for (int mt = 0; mt < 4; ++mt)
#pragma unroll
      for (int i2 = 0; i2 < 8; ++i2) {
        const int row = mt * 16 + i2 + 8 * h2;
        if (row < NTOK)
          out[((size_t)b * NTOK + row) * D_MODEL + c] = acc[mt][i2] + bc;
      }
  }
}

// ---------------------------------------------------------------------------
extern "C" void kernel_launch(void* const* d_in, const int* in_sizes, int n_in,
                              void* d_out, int out_size, void* d_ws, size_t ws_size,
                              hipStream_t stream) {
  const float* x    = (const float*)d_in[0];
  const float* mask = (const float*)d_in[1];
  const float* Wq   = (const float*)d_in[2];
  const float* bq   = (const float*)d_in[3];
  const float* Wk   = (const float*)d_in[4];
  const float* bk   = (const float*)d_in[5];
  const float* Wv   = (const float*)d_in[6];
  const float* bv   = (const float*)d_in[7];
  const float* Wp   = (const float*)d_in[8];
  const float* bp   = (const float*)d_in[9];
  float* out = (float*)d_out;

  const int B = in_sizes[0] / (NTOK * D_MODEL);   // 4096

  // workspace layout (halves):
  //   wbuf [4][384][384] (transposed f16 weights)
  //   qbuf [B*H][64][32] | kbuf same | vtbuf [B*H][32][64] | obuf [B][49][384]
  const size_t wElems   = (size_t)4 * D_MODEL * D_MODEL;
  const size_t qkvElems = (size_t)B * NH * NPAD * HDIM;
  _Float16* wbuf  = (_Float16*)d_ws;
  _Float16* qbuf  = wbuf + wElems;
  _Float16* kbuf  = qbuf + qkvElems;
  _Float16* vtbuf = kbuf + qkvElems;
  _Float16* obuf  = vtbuf + qkvElems;

  cvt_weights_kernel<<<dim3((D_MODEL * D_MODEL + 255) / 256, 4), 256, 0, stream>>>(
      Wq, Wk, Wv, Wp, wbuf);
  qkv_proj_kernel<<<dim3(B, 3), 256, 0, stream>>>(
      x, wbuf, bq, bk, bv, qbuf, kbuf, vtbuf);
  attn_kernel<<<B * NH, 128, 0, stream>>>(qbuf, kbuf, vtbuf, mask, obuf);
  out_proj_kernel<<<B, 256, 0, stream>>>(obuf, wbuf, bp, out);
}
